// GAT_47502338293966
// MI455X (gfx1250) — matrix operand order repaired and checked by reading, not compiled, not run on hardware
//
#include <hip/hip_runtime.h>
#include <hip/hip_bf16.h>

// ---------------------------------------------------------------------------
// 2-layer GAT on MI455X (gfx1250, wave32).
//   GEMMs: v_wmma_f32_16x16x32_f16, f16 staged in LDS, W tile moved by the
//          Tensor Data Mover (tensor_load_to_lds + s_wait_tensorcnt).
//   Edge softmax + aggregation: L2-resident float atomics (h1/out fit in the
//          192MB L2, so the 850k-edge gather/scatter runs at L2 bandwidth).
// ---------------------------------------------------------------------------

#define IN_CH  128
#define HID    32
#define HEADS  4
#define OUT_CH 64
#define NEG_SLOPE 0.2f

typedef __attribute__((ext_vector_type(16))) _Float16 v16h;
typedef __attribute__((ext_vector_type(8)))  float    v8f;

#if defined(__HIP_DEVICE_COMPILE__) && defined(__gfx1250__) && \
    __has_builtin(__builtin_amdgcn_tensor_load_to_lds)
#define USE_TDM 1
#else
#define USE_TDM 0
#endif

#if USE_TDM
typedef unsigned int u32x4 __attribute__((ext_vector_type(4)));
typedef int          i32x8 __attribute__((ext_vector_type(8)));
typedef int          i32x4 __attribute__((ext_vector_type(4)));

// 1-D contiguous global->LDS DMA of n f32 elements via the TDM.
// D# layout per CDNA5 ISA (groups 0/1; trailing groups 0).
// This toolchain's builtin is the 6-arg clang-23 form:
//   (uint32x4 g0, int32x8 g1, int32x4, int32x4, int32x8, i32 cpol)
__device__ inline void tdm_copy_f32_to_lds(const float* gsrc, unsigned lds_byte_off,
                                           unsigned n) {
    unsigned long long ga = (unsigned long long)(uintptr_t)gsrc;
    u32x4 g0;
    g0[0] = 1u;                                          // count=1 (valid user D#)
    g0[1] = lds_byte_off;                                // lds_addr
    g0[2] = (unsigned)ga;                                // global_addr[31:0]
    g0[3] = (unsigned)((ga >> 32) & 0x01FFFFFFull)       // global_addr[56:32]
          | (2u << 30);                                  // type=2 ("image")
    i32x8 g1;
    g1[0] = (int)(2u << 16);                             // data_size=4B, mask=0
    g1[1] = (int)((n & 0xFFFFu) << 16);                  // tensor_dim0[15:0]
    g1[2] = (int)(((n >> 16) & 0xFFFFu) | (1u << 16));   // tensor_dim0[31:16], dim1=1
    g1[3] = (int)((n & 0xFFFFu) << 16);                  // tile_dim0 = n (dim1 hi = 0)
    g1[4] = 1;                                           // tile_dim1=1, tile_dim2=0
    g1[5] = (int)n;                                      // tensor_dim0_stride = n
    g1[6] = 0;
    g1[7] = 0;
    i32x4 gz4 = {0, 0, 0, 0};
    i32x8 gz8 = {0, 0, 0, 0, 0, 0, 0, 0};
    __builtin_amdgcn_tensor_load_to_lds(g0, g1, gz4, gz4, gz8, 0);
}
#endif

// ---------------------------------------------------------------- utilities
__device__ inline float atomicMaxFloat(float* addr, float value) {
    if (value >= 0.0f)
        return __int_as_float(atomicMax((int*)addr, __float_as_int(value)));
    else
        return __uint_as_float(atomicMin((unsigned int*)addr, __float_as_uint(value)));
}

__global__ void k_fill(float* __restrict__ p, float v, long n) {
    long i = (long)blockIdx.x * blockDim.x + threadIdx.x;
    if (i < n) p[i] = v;
}

__global__ void k_init_bias(float* __restrict__ p, const float* __restrict__ b,
                            int C, long n) {
    long i = (long)blockIdx.x * blockDim.x + threadIdx.x;
    if (i < n) p[i] = b[i % C];
}

__global__ void k_bias_elu(float* __restrict__ x, const float* __restrict__ b,
                           int C, long n) {
    long i = (long)blockIdx.x * blockDim.x + threadIdx.x;
    if (i < n) {
        float v = x[i] + b[i % C];
        x[i] = v > 0.0f ? v : (expf(v) - 1.0f);
    }
}

// ------------------------------------------------------- WMMA GEMM (K=128)
// C[nrows, NOUT] = A[nrows,128] (f32->f16) x W[128,NOUT] (f32->f16)
// ROWT 16-row tiles per block (nrows must be a multiple of 16*ROWT),
// one 16-col tile per wave; B fragments held in registers across row tiles.
template<int NOUT, int ROWT>
__global__ __launch_bounds__((NOUT / 16) * 32)
void k_gemm_wmma(const float* __restrict__ A, const float* __restrict__ W,
                 float* __restrict__ C, int nrows) {
    constexpr int K    = 128;
    constexpr int NW   = NOUT / 16;
    constexpr int ROWS = 16 * ROWT;
    __shared__ _Float16 ldsA[ROWS * K];    // row-major [m][k]
    __shared__ _Float16 ldsWT[NOUT * K];   // transposed [n][k] -> K contiguous
#if USE_TDM
    __shared__ float ldsWf32[NOUT * K];    // raw f32 W tile landed by the TDM
#endif

    const int  t       = threadIdx.x;
    const int  nth     = NW * 32;
    const long rowbase = (long)blockIdx.x * ROWS;

#if USE_TDM
    if (t < 32)  // wave 0 kicks the DMA; EXEC is ignored by tensor ops
        tdm_copy_f32_to_lds(W, (unsigned)(uintptr_t)&ldsWf32[0], NOUT * K);
#endif

    // stage A tile (f32 global -> f16 LDS); A rows are contiguous, flat copy
    for (int idx = t; idx < ROWS * K; idx += nth)
        ldsA[idx] = (_Float16)A[rowbase * K + idx];

#if USE_TDM
    __syncthreads();                                   // A staged
    if (t < 32) __builtin_amdgcn_s_wait_tensorcnt(0);  // W tile landed
    __syncthreads();                                   // ... visible to all waves
    for (int idx = t; idx < NOUT * K; idx += nth) {
        int k = idx / NOUT, n = idx % NOUT;
        ldsWT[n * K + k] = (_Float16)ldsWf32[idx];
    }
#else
    for (int idx = t; idx < NOUT * K; idx += nth) {
        int k = idx / NOUT, n = idx % NOUT;
        ldsWT[n * K + k] = (_Float16)W[k * NOUT + n];
    }
#endif
    __syncthreads();

    const int w     = t >> 5;
    const int lane  = t & 31;
    const int cb    = w * 16;
    const int mlane = lane & 15;          // A row within tile
    const int kbA   = (lane >> 4) * 16;   // A K sub-base (contiguous, matches B)
    const int kbB   = (lane >> 4) * 16;   // B K sub-base
    const int nlane = lane & 15;          // B/C column
    const int mbase = (lane >> 4) * 8;    // C/D row sub-base

    union F { v16h v; unsigned u[8]; };

    // B fragments: resident in registers for all K chunks
    F bf[K / 32];
#pragma unroll
    for (int kk = 0; kk < K / 32; ++kk)
#pragma unroll
        for (int i = 0; i < 8; ++i)
            bf[kk].u[i] = *(const unsigned*)&ldsWT[(cb + nlane) * K + kk * 32 + kbB + 2 * i];

#pragma unroll
    for (int rt = 0; rt < ROWT; ++rt) {
        v8f acc = {};
        F af;
#pragma unroll
        for (int kk = 0; kk < K / 32; ++kk) {
#pragma unroll
            for (int i = 0; i < 8; ++i) {
                int ka = kk * 32 + kbA + 2 * i;
                af.u[i] = *(const unsigned*)&ldsA[(rt * 16 + mlane) * K + ka];
            }
            acc = __builtin_amdgcn_wmma_f32_16x16x32_f16(
                false, af.v, false, bf[kk].v, (short)0, acc, false, false);
        }
        // C/D layout: vgpr r -> M = r + 8*(lane>=16), N = lane&15
#pragma unroll
        for (int r = 0; r < 8; ++r)
            C[(rowbase + rt * 16 + mbase + r) * NOUT + cb + nlane] = acc[r];
    }
}

// -------------------------------------------------- per-node attention dots
template<int H, int C>
__global__ void k_alpha(const float* __restrict__ hfeat,
                        const float* __restrict__ att_s,
                        const float* __restrict__ att_d,
                        float* __restrict__ as, float* __restrict__ ad) {
    const int n = blockIdx.x;
    const int w = threadIdx.x >> 5;   // head
    const int lane = threadIdx.x & 31;
    constexpr int TOT = H * C;
    float ps = 0.0f, pd = 0.0f;
    for (int c = lane; c < C; c += 32) {
        float hv = hfeat[(long)n * TOT + w * C + c];
        ps += hv * att_s[w * C + c];
        pd += hv * att_d[w * C + c];
    }
#pragma unroll
    for (int off = 16; off; off >>= 1) {
        ps += __shfl_xor(ps, off, 32);
        pd += __shfl_xor(pd, off, 32);
    }
    if (lane == 0) { as[n * H + w] = ps; ad[n * H + w] = pd; }
}

// ------------------------------------------------------------- edge passes
__device__ inline void edge_endpoints(int e, int E, const int* __restrict__ src,
                                      const int* __restrict__ dst, int& s, int& d) {
    if (e < E) { s = src[e]; d = dst[e]; }
    else       { s = e - E; d = e - E; }     // self-loops appended (PyG default)
}

template<int H>
__global__ void k_edge_max(const int* __restrict__ src, const int* __restrict__ dst,
                           int E, int ET,
                           const float* __restrict__ as, const float* __restrict__ ad,
                           float* __restrict__ emax) {
    int e = blockIdx.x * blockDim.x + threadIdx.x;
    if (e >= ET) return;
    int s, d; edge_endpoints(e, E, src, dst, s, d);
#pragma unroll
    for (int h = 0; h < H; ++h) {
        float v = as[s * H + h] + ad[d * H + h];
        v = v > 0.0f ? v : NEG_SLOPE * v;           // leaky_relu
        atomicMaxFloat(&emax[d * H + h], v);
    }
}

template<int H>
__global__ void k_edge_denom(const int* __restrict__ src, const int* __restrict__ dst,
                             int E, int ET,
                             const float* __restrict__ as, const float* __restrict__ ad,
                             const float* __restrict__ emax, float* __restrict__ denom) {
    int e = blockIdx.x * blockDim.x + threadIdx.x;
    if (e >= ET) return;
    int s, d; edge_endpoints(e, E, src, dst, s, d);
#pragma unroll
    for (int h = 0; h < H; ++h) {
        float v = as[s * H + h] + ad[d * H + h];
        v = v > 0.0f ? v : NEG_SLOPE * v;
        atomicAdd(&denom[d * H + h], expf(v - emax[d * H + h]));
    }
}

// one wave per edge: b128 gather of h[src] (L2-resident), scale, scatter-add.
template<int H, int C>
__global__ void k_edge_aggr(const int* __restrict__ src, const int* __restrict__ dst,
                            int E, int ET,
                            const float* __restrict__ as, const float* __restrict__ ad,
                            const float* __restrict__ emax, const float* __restrict__ denom,
                            const float* __restrict__ hfeat, float* __restrict__ out) {
    constexpr int TOT = H * C;
    constexpr int VEC = TOT / 32;                    // 4 (layer1) or 2 (layer2)
    int wave = (int)((blockIdx.x * (long)blockDim.x + threadIdx.x) >> 5);
    int lane = threadIdx.x & 31;
    if (wave >= ET) return;
    int s, d; edge_endpoints(wave, E, src, dst, s, d);

    __builtin_prefetch(&hfeat[(long)s * TOT], 0, 0); // global_prefetch_b8

    float alpha[H];
#pragma unroll
    for (int h = 0; h < H; ++h) {
        float v = as[s * H + h] + ad[d * H + h];
        v = v > 0.0f ? v : NEG_SLOPE * v;
        float ex = expf(v - emax[d * H + h]);
        alpha[h] = ex / (denom[d * H + h] + 1e-16f);
    }

    const int cbase = lane * VEC;
    const float a = alpha[cbase / C];                // VEC block never crosses a head
    float hv[VEC];
    if constexpr (VEC == 4) {
        float4 v4 = *(const float4*)(hfeat + (long)s * TOT + cbase);
        hv[0] = v4.x; hv[1] = v4.y; hv[2] = v4.z; hv[3] = v4.w;
    } else {
        float2 v2 = *(const float2*)(hfeat + (long)s * TOT + cbase);
        hv[0] = v2.x; hv[1] = v2.y;
    }
#pragma unroll
    for (int j = 0; j < VEC; ++j)
        atomicAdd(&out[(long)d * TOT + cbase + j], hv[j] * a);
}

// ---------------------------------------------------------------------------
extern "C" void kernel_launch(void* const* d_in, const int* in_sizes, int n_in,
                              void* d_out, int out_size, void* d_ws, size_t ws_size,
                              hipStream_t stream) {
    const float* x    = (const float*)d_in[0];
    const int*   ei   = (const int*)d_in[1];     // [2,E] (JAX default x32 -> int32)
    const float* W1   = (const float*)d_in[2];
    const float* as1w = (const float*)d_in[3];
    const float* ad1w = (const float*)d_in[4];
    const float* b1   = (const float*)d_in[5];
    const float* W2   = (const float*)d_in[6];
    const float* as2w = (const float*)d_in[7];
    const float* ad2w = (const float*)d_in[8];
    const float* b2   = (const float*)d_in[9];
    float* out = (float*)d_out;

    const int N  = in_sizes[0] / IN_CH;          // 50000 (= 625 * 80)
    const int E  = in_sizes[1] / 2;              // 800000
    const int ET = E + N;                        // + self-loops
    const int* srcI = ei;
    const int* dstI = ei + E;

    // workspace carve-out
    char* ws = (char*)d_ws;
    size_t off = 0;
    auto alloc = [&](size_t bytes) { char* p = ws + off; off = (off + bytes + 255) & ~(size_t)255; return p; };
    float* h1     = (float*)alloc((size_t)N * 128 * 4);  // x @ W1
    float* out1   = (float*)alloc((size_t)N * 128 * 4);  // layer-1 aggregate / act
    float* h2     = (float*)alloc((size_t)N * 64 * 4);   // act1 @ W2
    float* alphaS1= (float*)alloc((size_t)N * 4 * 4);
    float* alphaD1= (float*)alloc((size_t)N * 4 * 4);
    float* emax1  = (float*)alloc((size_t)N * 4 * 4);
    float* denom1 = (float*)alloc((size_t)N * 4 * 4);
    float* alphaS2= (float*)alloc((size_t)N * 4);
    float* alphaD2= (float*)alloc((size_t)N * 4);
    float* emax2  = (float*)alloc((size_t)N * 4);
    float* denom2 = (float*)alloc((size_t)N * 4);
    (void)ws_size;

    constexpr int ROWT = 5;                       // 80 rows/block; 50000 % 80 == 0
    const int rowBlocks = (N + 16 * ROWT - 1) / (16 * ROWT);
    const int TB = 256;
    auto gridN = [&](long n) { return dim3((unsigned)((n + TB - 1) / TB)); };

    // ---------------- layer 1 ----------------
    k_gemm_wmma<128, ROWT><<<rowBlocks, 256, 0, stream>>>(x, W1, h1, N);
    k_alpha<HEADS, HID><<<N, HEADS * 32, 0, stream>>>(h1, as1w, ad1w, alphaS1, alphaD1);

    k_fill<<<gridN((long)N * 4), TB, 0, stream>>>(emax1, -3.402823466e38f, (long)N * 4);
    k_fill<<<gridN((long)N * 4), TB, 0, stream>>>(denom1, 0.0f, (long)N * 4);
    k_fill<<<gridN((long)N * 128), TB, 0, stream>>>(out1, 0.0f, (long)N * 128);

    k_edge_max<HEADS><<<gridN(ET), TB, 0, stream>>>(srcI, dstI, E, ET, alphaS1, alphaD1, emax1);
    k_edge_denom<HEADS><<<gridN(ET), TB, 0, stream>>>(srcI, dstI, E, ET, alphaS1, alphaD1, emax1, denom1);
    k_edge_aggr<HEADS, HID><<<dim3((unsigned)((ET + 7) / 8)), TB, 0, stream>>>(
        srcI, dstI, E, ET, alphaS1, alphaD1, emax1, denom1, h1, out1);

    k_bias_elu<<<gridN((long)N * 128), TB, 0, stream>>>(out1, b1, 128, (long)N * 128);

    // ---------------- layer 2 ----------------
    k_gemm_wmma<64, ROWT><<<rowBlocks, 128, 0, stream>>>(out1, W2, h2, N);
    k_alpha<1, OUT_CH><<<N, 32, 0, stream>>>(h2, as2w, ad2w, alphaS2, alphaD2);

    k_fill<<<gridN(N), TB, 0, stream>>>(emax2, -3.402823466e38f, N);
    k_fill<<<gridN(N), TB, 0, stream>>>(denom2, 0.0f, N);
    k_init_bias<<<gridN((long)N * 64), TB, 0, stream>>>(out, b2, 64, (long)N * 64);

    k_edge_max<1><<<gridN(ET), TB, 0, stream>>>(srcI, dstI, E, ET, alphaS2, alphaD2, emax2);
    k_edge_denom<1><<<gridN(ET), TB, 0, stream>>>(srcI, dstI, E, ET, alphaS2, alphaD2, emax2, denom2);
    k_edge_aggr<1, OUT_CH><<<dim3((unsigned)((ET + 7) / 8)), TB, 0, stream>>>(
        srcI, dstI, E, ET, alphaS2, alphaD2, emax2, denom2, h2, out);
}
